// SlidingWindowAttention_32126355374401
// MI455X (gfx1250) — compile-verified, matrix operand order checked by reading, not run on hardware
//
#include <hip/hip_runtime.h>
#include <hip/hip_bf16.h>
#include <math.h>

// Sliding-window attention, bf16 WMMA pipeline for gfx1250 (MI455X).
// Stages: cast->bf16, fused QKV GEMM (TDM-staged LDS + WMMA), dwconv,
// rope+l2norm+pack, flash sliding-window attention (WMMA), output GEMM.

#define B_    2
#define S_    2048
#define H_    16
#define D_    64
#define C_    1024
#define WIN_  512
#define SCALE_ 0.125f   // 64^-0.5

typedef __attribute__((ext_vector_type(16))) __bf16 v16bf;
typedef __attribute__((ext_vector_type(8)))  __bf16 v8bf;
typedef __attribute__((ext_vector_type(8)))  float  v8f;

#if defined(__has_builtin)
#  if __has_builtin(__builtin_amdgcn_tensor_load_to_lds)
#    define USE_TDM 1
#  endif
#endif
#ifndef USE_TDM
#  define USE_TDM 0
#endif

__device__ __forceinline__ v8f zero8() {
  v8f z;
#pragma unroll
  for (int i = 0; i < 8; ++i) z[i] = 0.0f;
  return z;
}

// Load a 16x32 bf16 fragment (A-layout / mirrored B-layout) from a row-major
// matrix with leading dimension ld (global or LDS; addrspace inferred after
// inlining). Per ISA 16-bit A layout (wave32): lane&15 = row, lane>>4 selects
// K-halves; two contiguous 16-byte loads per lane.
__device__ __forceinline__ v16bf load_frag16x32(const __bf16* __restrict__ base,
                                                int ld, int row0, int k0,
                                                int lane, int maxrow) {
  int r = row0 + (lane & 15);
  r = (r < maxrow) ? r : maxrow;
  int kk = k0 + ((lane >> 4) << 3);
  const __bf16* p = base + (size_t)r * ld + kk;
  v8bf lo = *(const v8bf*)(p);
  v8bf hi = *(const v8bf*)(p + 16);
  v16bf f;
#pragma unroll
  for (int i = 0; i < 8; ++i) { f[i] = lo[i]; f[i + 8] = hi[i]; }
  return f;
}

#if USE_TDM
typedef unsigned int v4u __attribute__((ext_vector_type(4)));
typedef int          v4i __attribute__((ext_vector_type(4)));
typedef int          v8i __attribute__((ext_vector_type(8)));

// Issue one TDM 2D tile load: global (row-major, ld elems/row, bf16) -> LDS.
// D# per CDNA5 ISA:
//   g0: count=1 | lds_addr[63:32] | global_addr[120:64] | type=2[127:126]
//   g1: data_size=1(2B) | tensor_dim0=ld | tensor_dim1=rows_total |
//       tile_dim0=cols | tile_dim1=rows | tensor_dim0_stride=ld
__device__ __forceinline__ void tdm_load_tile_2d(unsigned lds_off,
                                                 const __bf16* g,
                                                 int tile_rows, int tile_cols,
                                                 int tensor_rows, int ld) {
  unsigned long long ga = (unsigned long long)(size_t)g;
  v4u g0;
  g0[0] = 1u;                                   // count=1 (valid), load, no gather
  g0[1] = lds_off;                              // lds_addr (bytes)
  g0[2] = (unsigned)ga;                         // global_addr[31:0]
  g0[3] = (unsigned)((ga >> 32) & 0x01ffffffu)  // global_addr[56:32]
        | (2u << 30);                           // type=2 ("image")
  v8i g1;
  g1[0] = (int)(1u << 16);                      // wg_mask=0, data_size=1 (2 bytes)
  g1[1] = (int)(((unsigned)ld & 0xffffu) << 16);                    // tdim0 lo16
  g1[2] = (int)((((unsigned)ld >> 16) & 0xffffu)
        | (((unsigned)tensor_rows & 0xffffu) << 16));               // tdim0 hi | tdim1 lo
  g1[3] = (int)((((unsigned)tensor_rows >> 16) & 0xffffu)
        | (((unsigned)tile_cols & 0xffffu) << 16));                 // tdim1 hi | tile_dim0
  g1[4] = (int)((unsigned)tile_rows & 0xffffu);                     // tile_dim1, tile_dim2=0
  g1[5] = (int)(unsigned)ld;                                        // tdim0_stride[31:0]
  g1[6] = 0;                                                        // stride hi, tdim1_stride lo
  g1[7] = 0;
  v4i z4; z4[0] = 0; z4[1] = 0; z4[2] = 0; z4[3] = 0;
#if defined(__clang_major__) && (__clang_major__ >= 23)
  v8i z8;
#pragma unroll
  for (int i = 0; i < 8; ++i) z8[i] = 0;
  __builtin_amdgcn_tensor_load_to_lds(g0, g1, z4, z4, z8, 0);
#else
  __builtin_amdgcn_tensor_load_to_lds(g0, g1, z4, z4, 0);
#endif
}
#endif  // USE_TDM

// Fallback cooperative tile stage (global -> LDS) with b128 traffic.
__device__ __forceinline__ void stage_tile_coop(__bf16* dst, const __bf16* src,
                                                int rows, int ld, int k0, int tid) {
  const int chunks = rows * 8;  // 8 x (8 bf16 = 16B) chunks per 64-col row
  for (int i = tid; i < chunks; i += 256) {
    int r = i >> 3;
    int c = (i & 7) * 8;
    *(v8bf*)(dst + r * 64 + c) = *(const v8bf*)(src + (size_t)r * ld + k0 + c);
  }
}

__global__ void cast_f32_bf16(const float* __restrict__ in,
                              __bf16* __restrict__ out, int n) {
  int i = blockIdx.x * blockDim.x + threadIdx.x;
  if (i < n) out[i] = (__bf16)in[i];
}

// C[M,N] = A[M,K] @ B[N,K]^T + bias   (A,B bf16 row-major, C f32)
// Block: 256 threads = 8 waves (2x4); block tile 64x128; K-tile 64.
// Tiles staged into double-buffered LDS by the Tensor Data Mover (wave 0
// issues descriptors, TENSORcnt + barrier publishes); WMMA consumes via
// ds_load_b128 fragment reads.
__global__ __launch_bounds__(256)
void gemm_bf16_wmma(const __bf16* __restrict__ A, const __bf16* __restrict__ Bm,
                    const float* __restrict__ bias, float* __restrict__ Cout,
                    int M, int N, int K) {
  __shared__ __bf16 Asb[2][64][64];    // 2 x 8 KB
  __shared__ __bf16 Bsb[2][128][64];   // 2 x 16 KB
  const int lane = threadIdx.x & 31;
  const int wave = threadIdx.x >> 5;
  const int wm = wave >> 2, wn = wave & 3;
  const int rowblk = blockIdx.x * 64;
  const int colblk = blockIdx.y * 128;

  v8f c00 = zero8(), c01 = zero8(), c10 = zero8(), c11 = zero8();

  // ---- stage K-tile 0 into buffer 0 ----
#if USE_TDM
  if (threadIdx.x < 32) {
    tdm_load_tile_2d((unsigned)(size_t)(void*)&Asb[0][0][0],
                     A + (size_t)rowblk * K, 64, 64, M, K);
    tdm_load_tile_2d((unsigned)(size_t)(void*)&Bsb[0][0][0],
                     Bm + (size_t)colblk * K, 128, 64, N, K);
    __builtin_amdgcn_s_wait_tensorcnt(0);
  }
  __syncthreads();
#else
  stage_tile_coop(&Asb[0][0][0], A + (size_t)rowblk * K, 64, K, 0, threadIdx.x);
  stage_tile_coop(&Bsb[0][0][0], Bm + (size_t)colblk * K, 128, K, 0, threadIdx.x);
  __syncthreads();
#endif

  int cur = 0;
  for (int k0 = 0; k0 < K; k0 += 64) {
    const int knext = k0 + 64;
    // ---- async-stage next K-tile into the other buffer ----
    if (knext < K) {
#if USE_TDM
      if (threadIdx.x < 32) {
        tdm_load_tile_2d((unsigned)(size_t)(void*)&Asb[cur ^ 1][0][0],
                         A + (size_t)rowblk * K + knext, 64, 64, M, K);
        tdm_load_tile_2d((unsigned)(size_t)(void*)&Bsb[cur ^ 1][0][0],
                         Bm + (size_t)colblk * K + knext, 128, 64, N, K);
      }
#else
      stage_tile_coop(&Asb[cur ^ 1][0][0], A + (size_t)rowblk * K, 64, K, knext,
                      threadIdx.x);
      stage_tile_coop(&Bsb[cur ^ 1][0][0], Bm + (size_t)colblk * K, 128, K, knext,
                      threadIdx.x);
#endif
    }
    // ---- compute two K=32 WMMA steps from buffer `cur` ----
#pragma unroll
    for (int ks = 0; ks < 64; ks += 32) {
      v16bf a0 = load_frag16x32(&Asb[cur][0][0], 64, wm * 32,      ks, lane, 63);
      v16bf a1 = load_frag16x32(&Asb[cur][0][0], 64, wm * 32 + 16, ks, lane, 63);
      v16bf b0 = load_frag16x32(&Bsb[cur][0][0], 64, wn * 32,      ks, lane, 127);
      v16bf b1 = load_frag16x32(&Bsb[cur][0][0], 64, wn * 32 + 16, ks, lane, 127);
      c00 = __builtin_amdgcn_wmma_f32_16x16x32_bf16(false, a0, false, b0, (short)0, c00, false, false);
      c01 = __builtin_amdgcn_wmma_f32_16x16x32_bf16(false, a0, false, b1, (short)0, c01, false, false);
      c10 = __builtin_amdgcn_wmma_f32_16x16x32_bf16(false, a1, false, b0, (short)0, c10, false, false);
      c11 = __builtin_amdgcn_wmma_f32_16x16x32_bf16(false, a1, false, b1, (short)0, c11, false, false);
    }
    __syncthreads();  // all waves done reading buffer `cur`
    if (knext < K) {
#if USE_TDM
      if (threadIdx.x < 32) __builtin_amdgcn_s_wait_tensorcnt(0);
#endif
      __syncthreads();  // next buffer published to all waves
      cur ^= 1;
    }
  }

  const int half = lane >> 4, nl = lane & 15;
  const int row0 = rowblk + wm * 32;
  const int col0 = colblk + wn * 32;
#pragma unroll
  for (int r = 0; r < 8; ++r) {
    int m0 = row0 + r + 8 * half;
    int m1 = m0 + 16;
    int n0 = col0 + nl, n1 = n0 + 16;
    Cout[(size_t)m0 * N + n0] = c00[r] + bias[n0];
    Cout[(size_t)m0 * N + n1] = c01[r] + bias[n1];
    Cout[(size_t)m1 * N + n0] = c10[r] + bias[n0];
    Cout[(size_t)m1 * N + n1] = c11[r] + bias[n1];
  }
}

// Causal depthwise conv (KSIZE=4) on packed qkv (rows B*S, cols 3*C).
__global__ void dwconv3(const float* __restrict__ qkv,
                        const float* __restrict__ cqw, const float* __restrict__ cqb,
                        const float* __restrict__ ckw, const float* __restrict__ ckb,
                        const float* __restrict__ cvw, const float* __restrict__ cvb,
                        float* __restrict__ qc, float* __restrict__ kc,
                        float* __restrict__ vc) {
  int idx = blockIdx.x * blockDim.x + threadIdx.x;
  const int NT = B_ * S_ * C_;
  if (idx >= 3 * NT) return;
  int t = idx / NT;
  int rem = idx - t * NT;
  int b = rem / (S_ * C_);
  int rem2 = rem - b * (S_ * C_);
  int s = rem2 / C_;
  int c = rem2 - s * C_;
  const float* w  = (t == 0) ? cqw : (t == 1) ? ckw : cvw;
  const float* bb = (t == 0) ? cqb : (t == 1) ? ckb : cvb;
  float acc = bb[c];
#pragma unroll
  for (int j = 0; j < 4; ++j) {
    int sp = s - 3 + j;
    if (sp >= 0)
      acc += qkv[((size_t)(b * S_ + sp)) * (3 * C_) + t * C_ + c] * w[c * 4 + j];
  }
  float* o = (t == 0) ? qc : (t == 1) ? kc : vc;
  o[((size_t)(b * S_ + s)) * C_ + c] = acc;
}

// Rotary + l2norm on q,k; repack q,k,v (B,S,C)f32 -> (B,H,S,D)bf16.
__global__ void rope_norm_pack(const float* __restrict__ qc,
                               const float* __restrict__ kc,
                               const float* __restrict__ vc,
                               __bf16* __restrict__ qn, __bf16* __restrict__ kn,
                               __bf16* __restrict__ vn) {
  int idx = blockIdx.x * blockDim.x + threadIdx.x;  // over B*H*S
  if (idx >= B_ * H_ * S_) return;
  int b = idx / (H_ * S_);
  int rem = idx - b * H_ * S_;
  int h = rem / S_;
  int s = rem - h * S_;
  const size_t src = ((size_t)(b * S_ + s)) * C_ + h * D_;
  const size_t dst = ((size_t)((b * H_ + h) * S_ + s)) * D_;

  float cs[32], sn[32];
#pragma unroll
  for (int i = 0; i < 32; ++i) {
    // inv_freq = 10000^(-2i/64); ln(10000)=9.21034...
    float fr = __expf(-(float)i * (9.2103403719761827f / 32.0f)) * (float)s;
    cs[i] = __cosf(fr);
    sn[i] = __sinf(fr);
  }

  float buf[64];
  // q: rotary + l2norm
  for (int d = 0; d < 64; ++d) buf[d] = qc[src + d];
  float ss = 1e-12f;
#pragma unroll
  for (int i = 0; i < 32; ++i) {
    float x1 = buf[i], x2 = buf[i + 32];
    float r1 = x1 * cs[i] - x2 * sn[i];
    float r2 = x2 * cs[i] + x1 * sn[i];
    buf[i] = r1; buf[i + 32] = r2;
    ss += r1 * r1 + r2 * r2;
  }
  float inv = rsqrtf(ss);
  for (int d = 0; d < 64; ++d) qn[dst + d] = (__bf16)(buf[d] * inv);

  // k: rotary + l2norm
  for (int d = 0; d < 64; ++d) buf[d] = kc[src + d];
  ss = 1e-12f;
#pragma unroll
  for (int i = 0; i < 32; ++i) {
    float x1 = buf[i], x2 = buf[i + 32];
    float r1 = x1 * cs[i] - x2 * sn[i];
    float r2 = x2 * cs[i] + x1 * sn[i];
    buf[i] = r1; buf[i + 32] = r2;
    ss += r1 * r1 + r2 * r2;
  }
  inv = rsqrtf(ss);
  for (int d = 0; d < 64; ++d) kn[dst + d] = (__bf16)(buf[d] * inv);

  // v: cast only
  for (int d = 0; d < 64; ++d) vn[dst + d] = (__bf16)vc[src + d];
}

// Flash-style sliding-window attention. One wave per 16-query tile; 32-key
// blocks; scores via 4 WMMAs (K=64), online softmax via wave32 shfl_xor in
// 16-lane groups (matches C/D half layout), P re-fragmented via LDS, PxV via
// 4 WMMAs. Output written as bf16 (B,S,C) for the Wo GEMM.
__global__ __launch_bounds__(128)
void attn_sliding_wmma(const __bf16* __restrict__ qn, const __bf16* __restrict__ kn,
                       const __bf16* __restrict__ vn, __bf16* __restrict__ attn_out) {
  __shared__ __bf16 Pl[4][16][32];  // per-wave 16x32 P tile
  const int lane = threadIdx.x & 31;
  const int wave = threadIdx.x >> 5;
  const int bh = blockIdx.y;  // b*H + h
  const int b = bh >> 4;
  const int h = bh & 15;
  const int q0 = (blockIdx.x * 4 + wave) * 16;
  const int half = lane >> 4, nl = lane & 15;

  const __bf16* qb = qn + (size_t)bh * S_ * D_;
  const __bf16* kb = kn + (size_t)bh * S_ * D_;
  const __bf16* vb = vn + (size_t)bh * S_ * D_;

  v16bf aq0 = load_frag16x32(qb, D_, q0, 0,  lane, S_ - 1);
  v16bf aq1 = load_frag16x32(qb, D_, q0, 32, lane, S_ - 1);

  float mrun[8], lrun[8];
  v8f acc[4];
#pragma unroll
  for (int r = 0; r < 8; ++r) { mrun[r] = -1e30f; lrun[r] = 0.0f; }
#pragma unroll
  for (int j = 0; j < 4; ++j) acc[j] = zero8();

  int kstart = q0 - (WIN_ - 1);
  if (kstart < 0) kstart = 0;
  kstart &= ~31;
  const int kend = q0 + 15;

  for (int k0 = kstart; k0 <= kend; k0 += 32) {
    v8f sc[2];
#pragma unroll
    for (int t = 0; t < 2; ++t) {
      v16bf bk0 = load_frag16x32(kb, D_, k0 + t * 16, 0,  lane, S_ - 1);
      v16bf bk1 = load_frag16x32(kb, D_, k0 + t * 16, 32, lane, S_ - 1);
      v8f z = zero8();
      z     = __builtin_amdgcn_wmma_f32_16x16x32_bf16(false, aq0, false, bk0, (short)0, z, false, false);
      sc[t] = __builtin_amdgcn_wmma_f32_16x16x32_bf16(false, aq1, false, bk1, (short)0, z, false, false);
    }

#pragma unroll
    for (int r = 0; r < 8; ++r) {
      const int m = q0 + r + 8 * half;
      int j0 = k0 + nl;
      int j1 = k0 + 16 + nl;
      bool ok0 = (j0 <= m) && (m - j0 < WIN_);
      bool ok1 = (j1 <= m) && (m - j1 < WIN_);
      float sv0 = ok0 ? sc[0][r] * SCALE_ : -1e30f;
      float sv1 = ok1 ? sc[1][r] * SCALE_ : -1e30f;
      float red = fmaxf(sv0, sv1);
#pragma unroll
      for (int d = 1; d < 16; d <<= 1) red = fmaxf(red, __shfl_xor(red, d, 32));
      float mnew = fmaxf(mrun[r], red);
      float corr = __expf(mrun[r] - mnew);  // finite sentinels: no NaN
      float p0 = ok0 ? __expf(sv0 - mnew) : 0.0f;
      float p1 = ok1 ? __expf(sv1 - mnew) : 0.0f;
      float rs = p0 + p1;
#pragma unroll
      for (int d = 1; d < 16; d <<= 1) rs += __shfl_xor(rs, d, 32);
      lrun[r] = lrun[r] * corr + rs;
      mrun[r] = mnew;
#pragma unroll
      for (int j = 0; j < 4; ++j) acc[j][r] *= corr;
      Pl[wave][r + 8 * half][nl]      = (__bf16)p0;
      Pl[wave][r + 8 * half][16 + nl] = (__bf16)p1;
    }

    // wave-local LDS RAW: DS ops are in-order per wave, wait before reuse
    asm volatile("s_wait_dscnt 0" ::: "memory");

    v16bf ap;
    {
      const __bf16* pr = &Pl[wave][nl][0];
      v8bf lo = *(const v8bf*)(pr + half * 8);
      v8bf hi = *(const v8bf*)(pr + 16 + half * 8);
#pragma unroll
      for (int i = 0; i < 8; ++i) { ap[i] = lo[i]; ap[i + 8] = hi[i]; }
    }

#pragma unroll
    for (int j = 0; j < 4; ++j) {
      v16bf bvf;
#pragma unroll
      for (int v = 0; v < 8; ++v) {
        int kk = ((v < 4) ? 0 : 16) + half * 8 + (v & 3) * 2;
        int r0 = k0 + kk;     if (r0 > S_ - 1) r0 = S_ - 1;  // masked lanes: p==0
        int r1 = k0 + kk + 1; if (r1 > S_ - 1) r1 = S_ - 1;
        bvf[2 * v]     = vb[(size_t)r0 * D_ + j * 16 + nl];
        bvf[2 * v + 1] = vb[(size_t)r1 * D_ + j * 16 + nl];
      }
      acc[j] = __builtin_amdgcn_wmma_f32_16x16x32_bf16(false, ap, false, bvf, (short)0, acc[j], false, false);
    }
    asm volatile("" ::: "memory");
  }

#pragma unroll
  for (int j = 0; j < 4; ++j) {
#pragma unroll
    for (int r = 0; r < 8; ++r) {
      int srow = q0 + r + 8 * half;
      int col = h * D_ + j * 16 + nl;
      float o = acc[j][r] / lrun[r];
      attn_out[((size_t)(b * S_ + srow)) * C_ + col] = (__bf16)o;
    }
  }
}

extern "C" void kernel_launch(void* const* d_in, const int* in_sizes, int n_in,
                              void* d_out, int out_size, void* d_ws, size_t ws_size,
                              hipStream_t stream) {
  (void)in_sizes; (void)n_in; (void)out_size; (void)ws_size;
  const float* x   = (const float*)d_in[0];
  const float* Wq  = (const float*)d_in[1];
  const float* bq  = (const float*)d_in[2];
  const float* Wk  = (const float*)d_in[3];
  const float* bk  = (const float*)d_in[4];
  const float* Wv  = (const float*)d_in[5];
  const float* bv  = (const float*)d_in[6];
  const float* cqw = (const float*)d_in[7];
  const float* cqb = (const float*)d_in[8];
  const float* ckw = (const float*)d_in[9];
  const float* ckb = (const float*)d_in[10];
  const float* cvw = (const float*)d_in[11];
  const float* cvb = (const float*)d_in[12];
  const float* Wo  = (const float*)d_in[13];
  const float* bo  = (const float*)d_in[14];
  float* out = (float*)d_out;

  char* ws = (char*)d_ws;
  // Workspace map (MB offsets); peak 137 MB, intermediates L2-resident.
  __bf16* xb    = (__bf16*)(ws);                               // x bf16      8 MB
  __bf16* wqkv  = (__bf16*)(ws + (size_t)(8)  * (1u << 20));   // Wq|Wk|Wv    6 MB
  __bf16* wo    = (__bf16*)(ws + (size_t)(14) * (1u << 20));   // Wo          2 MB
  float*  bqkv  = (float*) (ws + (size_t)(16) * (1u << 20));   // biases     12 KB
  float*  qkv   = (float*) (ws + (size_t)(17) * (1u << 20));   // qkv f32    48 MB
  float*  qc    = (float*) (ws + (size_t)(65) * (1u << 20));
  float*  kc    = (float*) (ws + (size_t)(81) * (1u << 20));
  float*  vc    = (float*) (ws + (size_t)(97) * (1u << 20));
  __bf16* qnb   = (__bf16*)(ws + (size_t)(113) * (1u << 20));
  __bf16* knb   = (__bf16*)(ws + (size_t)(121) * (1u << 20));
  __bf16* vnb   = (__bf16*)(ws + (size_t)(129) * (1u << 20));
  __bf16* attnb = (__bf16*)(ws + (size_t)(17)  * (1u << 20));  // reuse qkv (dead)

  const int T = 256;
  int n = B_ * S_ * C_;  // 4194304
  cast_f32_bf16<<<(n + T - 1) / T, T, 0, stream>>>(x, xb, n);
  n = C_ * C_;           // 1048576
  cast_f32_bf16<<<(n + T - 1) / T, T, 0, stream>>>(Wq, wqkv,         n);
  cast_f32_bf16<<<(n + T - 1) / T, T, 0, stream>>>(Wk, wqkv + n,     n);
  cast_f32_bf16<<<(n + T - 1) / T, T, 0, stream>>>(Wv, wqkv + 2 * n, n);
  cast_f32_bf16<<<(n + T - 1) / T, T, 0, stream>>>(Wo, wo,           n);
  hipMemcpyAsync(bqkv,          bq, C_ * sizeof(float), hipMemcpyDeviceToDevice, stream);
  hipMemcpyAsync(bqkv + C_,     bk, C_ * sizeof(float), hipMemcpyDeviceToDevice, stream);
  hipMemcpyAsync(bqkv + 2 * C_, bv, C_ * sizeof(float), hipMemcpyDeviceToDevice, stream);

  // Fused QKV projection: (4096x1024) @ (3072x1024)^T
  gemm_bf16_wmma<<<dim3((B_ * S_) / 64, (3 * C_) / 128), 256, 0, stream>>>(
      xb, wqkv, bqkv, qkv, B_ * S_, 3 * C_, C_);

  n = 3 * B_ * S_ * C_;
  dwconv3<<<(n + T - 1) / T, T, 0, stream>>>(qkv, cqw, cqb, ckw, ckb, cvw, cvb,
                                             qc, kc, vc);

  n = B_ * H_ * S_;  // 65536
  rope_norm_pack<<<(n + T - 1) / T, T, 0, stream>>>(qc, kc, vc, qnb, knb, vnb);

  attn_sliding_wmma<<<dim3(S_ / 64, B_ * H_), 128, 0, stream>>>(qnb, knb, vnb, attnb);

  // Output projection: (4096x1024) @ (1024x1024)^T + bo -> f32 out
  gemm_bf16_wmma<<<dim3((B_ * S_) / 64, C_ / 128), 256, 0, stream>>>(
      attnb, wo, bo, out, B_ * S_, C_, C_);
}
